// PLDecoder_41472204210715
// MI455X (gfx1250) — matrix operand order repaired and checked by reading, not compiled, not run on hardware
//
#include <hip/hip_runtime.h>

// ---------------------------------------------------------------------------
// PLDecoder layer for MI455X (gfx1250, wave32, WMMA).
// All GEMMs run on v_wmma_f32_16x16x32_f16 (f16 in, f32 accumulate).
// B fragments are register-resident per output-column tile; V is stored
// transposed in LDS so all WMMA operand loads are contiguous b128s.
// ---------------------------------------------------------------------------

typedef _Float16 half_t;
typedef __attribute__((ext_vector_type(16))) _Float16 v16h;
typedef __attribute__((ext_vector_type(8)))  float    v8f;

constexpr int kNQ = 20, kBS = 16, kNP = 72, kD = 256;
constexpr int kHEADS = 8, kHD = 32;
constexpr int kH = 64, kW = 160, kHW = kH * kW;
constexpr int kTOK = kNQ * kBS * kNP;   // 23040 tokens
constexpr int LDW  = 264;               // LDS row pitch (halfs)

// ---------------------------------------------------------------------------
// WMMA helpers (layouts per CDNA5 ISA 7.12.2, wave32)
// ---------------------------------------------------------------------------
__device__ __forceinline__ v8f wmma_ff(v16h a, v16h b, v8f c) {
  return __builtin_amdgcn_wmma_f32_16x16x32_f16(false, a, false, b, (short)0, c,
                                                false, false);
}

// A fragment (16x32) from row-major [M][ld] halfs.
// lane: m = lane&15; elems 0..7 -> k = hi*8 + j, elems 8..15 -> k = 16 + hi*8 + j.
__device__ __forceinline__ v16h load_A_rm(const half_t* X, int ld, int m0, int k0) {
  const int lane = threadIdx.x & 31;
  const half_t* p = X + (m0 + (lane & 15)) * ld + k0 + ((lane >> 4) * 8);
  v16h a;
#pragma unroll
  for (int j = 0; j < 8; ++j) { a[j] = p[j]; a[8 + j] = p[16 + j]; }
  return a;
}

// B fragment (32x16) where B[k][n] = W[n0+n][k0+k], W row-major [N][ld].
// lane: n = lane&15; elems 0..7 -> k = hi*8 + j, elems 8..15 -> k = 16 + hi*8 + j.
__device__ __forceinline__ v16h load_B_nk(const half_t* W, int ld, int n0, int k0) {
  const int lane = threadIdx.x & 31;
  const half_t* p = W + (long)(n0 + (lane & 15)) * ld + k0 + ((lane >> 4) * 8);
  v16h b;
#pragma unroll
  for (int j = 0; j < 8; ++j) { b[j] = p[j]; b[8 + j] = p[16 + j]; }
  return b;
}

// ---------------------------------------------------------------------------
// Weight conversion fp32 -> f16
// ---------------------------------------------------------------------------
__global__ void __launch_bounds__(256) cvt_f2h_kernel(const float* __restrict__ s,
                                                      half_t* __restrict__ d, int n) {
  int i = blockIdx.x * 256 + threadIdx.x;
  if (i < n) d[i] = (half_t)s[i];
}

// ---------------------------------------------------------------------------
// Fused MHA (8 heads, hd=32) + residual + LayerNorm. One batch element/block.
// Token s of batch bb lives at in[bb*baseStride + s*rowStride + d].
// LDS: Q[SPM][LDW] | K[SPK][LDW] | Vt[256][LDV] | X/O[SPM][LDW] | P[wave][16][SPK]
// ---------------------------------------------------------------------------
constexpr size_t mha_smem(int S) {
  int SPM = (S + 15) & ~15;
  int SPK = (S + 31) & ~31;
  int LDV = SPK + 8;
  return (size_t)((2 * SPM + SPK) * LDW + 256 * LDV + 8 * 16 * SPK) * sizeof(half_t);
}

template <int S>
__global__ void __launch_bounds__(256) mha_ln_kernel(
    const float* __restrict__ in, float* __restrict__ out,
    const half_t* __restrict__ w_in, const float* __restrict__ b_in,
    const half_t* __restrict__ w_out, const float* __restrict__ b_out,
    const float* __restrict__ ln_g, const float* __restrict__ ln_b,
    long rowStride, long baseStride) {
  constexpr int SPM = (S + 15) & ~15;   // query rows, padded to 16
  constexpr int SPK = (S + 31) & ~31;   // key rows, padded to 32 (for P@V K-dim)
  constexpr int LDV = SPK + 8;          // Vt row pitch
  constexpr int RT = SPM / 16;
  constexpr int KT = SPK / 16;
  constexpr int KT32 = SPK / 32;

  extern __shared__ char smem[];
  half_t* shQ  = (half_t*)smem;            // SPM x LDW
  half_t* shK  = shQ + SPM * LDW;          // SPK x LDW
  half_t* shVt = shK + SPK * LDW;          // 256 x LDV  (Vt[dim][token])
  half_t* shX  = shVt + 256 * LDV;         // SPM x LDW : X input, then O
  half_t* shP  = shX + SPM * LDW;          // per-wave 16 x SPK probs
  float*  shC  = (float*)shK;              // preLN reuse (SPM*LDW f32 fits K+Vt)

  const int tid = threadIdx.x;
  const int wave = tid >> 5;
  const int lane = tid & 31;
  const int ncol = lane & 15;
  const int hi = lane >> 4;
  const long base = (long)blockIdx.x * baseStride;

  { // zero Q,K,Vt,X (padding rows/cols must be finite zeros)
    unsigned* z = (unsigned*)smem;
    const int ztot = ((2 * SPM + SPK) * LDW + 256 * LDV) / 2;
    for (int i = tid; i < ztot; i += 256) z[i] = 0u;
  }
  __syncthreads();

  // load X (fp32 global) -> f16 LDS
  for (int i = tid; i < S * kD; i += 256) {
    int r = i >> 8, c = i & 255;
    shX[r * LDW + c] = (half_t)in[base + (long)r * rowStride + c];
  }
  __syncthreads();

  // ---- QKV projections: each wave owns (proj, nt) with B held in registers ----
  for (int c = wave; c < 3 * 16; c += 8) {
    int proj = c >> 4;
    int nt = c & 15;
    const half_t* Wp = w_in + proj * kD * kD;
    __builtin_prefetch(Wp + (long)(nt * 16) * kD, 0, 0);
    v16h bfrag[8];
#pragma unroll
    for (int kt = 0; kt < 8; ++kt) bfrag[kt] = load_B_nk(Wp, kD, nt * 16, kt * 32);
    int n = nt * 16 + ncol;
    float bias = b_in[proj * kD + n];
#pragma unroll
    for (int mt = 0; mt < RT; ++mt) {
      v8f acc = {};
#pragma unroll
      for (int kt = 0; kt < 8; ++kt)
        acc = wmma_ff(load_A_rm(shX, LDW, mt * 16, kt * 32), bfrag[kt], acc);
      int mb = mt * 16 + hi * 8;
      if (proj == 2) {          // V stored transposed: Vt[dim][token]
#pragma unroll
        for (int r = 0; r < 8; ++r)
          shVt[n * LDV + mb + r] = (half_t)(acc[r] + bias);
      } else {
        half_t* dst = (proj == 0) ? shQ : shK;
#pragma unroll
        for (int r = 0; r < 8; ++r)
          dst[(mb + r) * LDW + n] = (half_t)(acc[r] + bias);
      }
    }
  }
  __syncthreads();

  // ---- attention: per (head, row-tile) task ----
  const float scale = 0.17677669529663687f;  // 1/sqrt(32)
  half_t* myP = shP + wave * 16 * SPK;
  for (int t = wave; t < kHEADS * RT; t += 8) {
    int h = t / RT, mt = t - h * RT;
    v16h aq = load_A_rm(shQ, LDW, mt * 16, h * kHD);
    v8f sc[KT];
#pragma unroll
    for (int nt = 0; nt < KT; ++nt) {          // scores: hd=32 == one WMMA K-step
      v16h bk = load_B_nk(shK, LDW, nt * 16, h * kHD);
      v8f zz = {};
      sc[nt] = wmma_ff(aq, bk, zz);
    }
    // softmax over keys; each score row spans the 16 lanes sharing `hi`
#pragma unroll
    for (int r = 0; r < 8; ++r) {
      float mx = -1e30f;
#pragma unroll
      for (int nt = 0; nt < KT; ++nt) {
        int col = nt * 16 + ncol;
        float v = sc[nt][r] * scale;
        v = (col < S) ? v : -1e30f;
        sc[nt][r] = v;
        mx = fmaxf(mx, v);
      }
#pragma unroll
      for (int mk = 8; mk >= 1; mk >>= 1) mx = fmaxf(mx, __shfl_xor(mx, mk, 32));
      float sum = 0.f;
#pragma unroll
      for (int nt = 0; nt < KT; ++nt) {
        float e = __expf(sc[nt][r] - mx);
        sc[nt][r] = e;
        sum += e;
      }
#pragma unroll
      for (int mk = 8; mk >= 1; mk >>= 1) sum += __shfl_xor(sum, mk, 32);
      float inv = 1.f / sum;
#pragma unroll
      for (int nt = 0; nt < KT; ++nt) sc[nt][r] *= inv;
    }
    // C-layout -> A-layout via per-wave LDS staging
#pragma unroll
    for (int nt = 0; nt < KT; ++nt)
#pragma unroll
      for (int r = 0; r < 8; ++r)
        myP[(r + hi * 8) * SPK + nt * 16 + ncol] = (half_t)sc[nt][r];
    // O = P @ V : B[k][n] = Vt[dim n0+n][token k] -> contiguous b128 loads
    v8f o0 = {}, o1 = {};
#pragma unroll
    for (int kt = 0; kt < KT32; ++kt) {
      v16h ap = load_A_rm(myP, SPK, 0, kt * 32);
      o0 = wmma_ff(ap, load_B_nk(shVt, LDV, h * kHD, kt * 32), o0);
      o1 = wmma_ff(ap, load_B_nk(shVt, LDV, h * kHD + 16, kt * 32), o1);
    }
    int mb = mt * 16 + hi * 8;
#pragma unroll
    for (int r = 0; r < 8; ++r) {
      shX[(mb + r) * LDW + h * kHD + ncol] = (half_t)o0[r];
      shX[(mb + r) * LDW + h * kHD + 16 + ncol] = (half_t)o1[r];
    }
  }
  __syncthreads();

  // ---- out projection + bias + residual (B in registers, 2 col tiles/wave) ----
#pragma unroll
  for (int nt2 = 0; nt2 < 2; ++nt2) {
    int nt = wave * 2 + nt2;
    v16h bfrag[8];
#pragma unroll
    for (int kt = 0; kt < 8; ++kt) bfrag[kt] = load_B_nk(w_out, kD, nt * 16, kt * 32);
    int n = nt * 16 + ncol;
    float bias = b_out[n];
#pragma unroll
    for (int mt = 0; mt < RT; ++mt) {
      v8f acc = {};
#pragma unroll
      for (int kt = 0; kt < 8; ++kt)
        acc = wmma_ff(load_A_rm(shX, LDW, mt * 16, kt * 32), bfrag[kt], acc);
      int mb = mt * 16 + hi * 8;
#pragma unroll
      for (int r = 0; r < 8; ++r) {
        int m = mb + r;
        float resid = (m < S) ? in[base + (long)m * rowStride + n] : 0.f;
        shC[m * LDW + n] = acc[r] + bias + resid;
      }
    }
  }
  __syncthreads();

  // ---- LayerNorm (one row per wave iteration) ----
  for (int m = wave; m < S; m += 8) {
    float vals[8], s0 = 0.f, s1 = 0.f;
#pragma unroll
    for (int j = 0; j < 8; ++j) {
      float v = shC[m * LDW + lane * 8 + j];
      vals[j] = v; s0 += v; s1 += v * v;
    }
#pragma unroll
    for (int mk = 16; mk >= 1; mk >>= 1) {
      s0 += __shfl_xor(s0, mk, 32);
      s1 += __shfl_xor(s1, mk, 32);
    }
    float mean = s0 * (1.f / 256.f);
    float var = s1 * (1.f / 256.f) - mean * mean;
    float inv = rsqrtf(var + 1e-5f);
#pragma unroll
    for (int j = 0; j < 8; ++j) {
      int c = lane * 8 + j;
      out[base + (long)m * rowStride + c] = (vals[j] - mean) * inv * ln_g[c] + ln_b[c];
    }
  }
}

// ---------------------------------------------------------------------------
// Generic WMMA GEMM: C[M][256] = act(A[M][256] @ W[256][256]^T + bias [+ resid])
// 128 rows per block, 8 waves; each wave owns 2 column tiles with all 8 B
// fragments register-resident and sweeps the 8 row tiles.
// ---------------------------------------------------------------------------
template <typename AT, bool RELU, bool RESID, typename OT>
__global__ void __launch_bounds__(256) gemm256_wmma(
    const AT* __restrict__ A, const half_t* __restrict__ Wt,
    const float* __restrict__ bias, const float* __restrict__ resid,
    OT* __restrict__ C, int M) {
  const int lane = threadIdx.x & 31;
  const int wave = threadIdx.x >> 5;
  const int ncol = lane & 15;
  const int hi = lane >> 4;
  const int m0blk = blockIdx.x * 128;
#pragma unroll
  for (int nt2 = 0; nt2 < 2; ++nt2) {
    int nt = wave * 2 + nt2;
    __builtin_prefetch(Wt + (long)(nt * 16) * kD, 0, 0);
    v16h bfrag[8];
#pragma unroll
    for (int kt = 0; kt < 8; ++kt) bfrag[kt] = load_B_nk(Wt, kD, nt * 16, kt * 32);
    int n = nt * 16 + ncol;
    float bv = bias[n];
    for (int mt = 0; mt < 8; ++mt) {
      int m0 = m0blk + mt * 16;
      if (m0 >= M) continue;
      int m = m0 + ncol;
      const bool mval = (m < M);
      const AT* arow = A + (long)(mval ? m : (M - 1)) * kD;
      v8f acc = {};
#pragma unroll
      for (int kt = 0; kt < 8; ++kt) {
        v16h a = {};
        if (mval) {
          const AT* p = arow + kt * 32 + hi * 8;
#pragma unroll
          for (int j = 0; j < 8; ++j) { a[j] = (half_t)p[j]; a[8 + j] = (half_t)p[16 + j]; }
        }
        acc = wmma_ff(a, bfrag[kt], acc);
      }
      int mb = m0 + hi * 8;
#pragma unroll
      for (int r = 0; r < 8; ++r) {
        int mr = mb + r;
        if (mr < M) {
          float v = acc[r] + bv;
          if (RESID) v += resid[(long)mr * kD + n];
          if (RELU) v = fmaxf(v, 0.f);
          C[(long)mr * kD + n] = (OT)v;
        }
      }
    }
  }
}

// ---------------------------------------------------------------------------
// Sampling offsets + attention weights: one query per wave.
// locaw layout per query: sx[4], sy[4], aw[4]  (sx/sy already in pixel coords)
// ---------------------------------------------------------------------------
__global__ void __launch_bounds__(256) offaw_kernel(
    const float* __restrict__ tgt2, const float* __restrict__ ref,
    const float* __restrict__ off_w, const float* __restrict__ off_b,
    const float* __restrict__ aw_w, const float* __restrict__ aw_b,
    float* __restrict__ locaw) {
  const int lane = threadIdx.x & 31;
  const int wave = threadIdx.x >> 5;
  int qi = blockIdx.x * 8 + wave;
  if (qi >= kTOK) return;
  int b = qi / (kNQ * kNP), qq = qi - b * (kNQ * kNP);
  int q = qq / kNP, p = qq - q * kNP;
  const float* xr = tgt2 + ((long)((q * kBS + b) * kNP + p)) * kD;
  float x[8];
#pragma unroll
  for (int j = 0; j < 8; ++j) x[j] = xr[lane * 8 + j];
  float res[12];
#pragma unroll
  for (int o = 0; o < 12; ++o) {
    const float* w = (o < 8) ? (off_w + o * kD) : (aw_w + (o - 8) * kD);
    float s = 0.f;
#pragma unroll
    for (int j = 0; j < 8; ++j) s += x[j] * w[lane * 8 + j];
#pragma unroll
    for (int mk = 16; mk >= 1; mk >>= 1) s += __shfl_xor(s, mk, 32);
    res[o] = s + ((o < 8) ? off_b[o] : aw_b[o - 8]);
  }
  if (lane == 0) {
    float mx = fmaxf(fmaxf(res[8], res[9]), fmaxf(res[10], res[11]));
    float e[4], sum = 0.f;
#pragma unroll
    for (int i = 0; i < 4; ++i) { e[i] = __expf(res[8 + i] - mx); sum += e[i]; }
    float rx = ref[((long)b * (kNQ * kNP) + qq) * 2 + 0];
    float ry = ref[((long)b * (kNQ * kNP) + qq) * 2 + 1];
    float* o = locaw + (long)qi * 12;
#pragma unroll
    for (int i = 0; i < 4; ++i) {
      float lx = rx + res[2 * i]     / (float)kW;
      float ly = ry + res[2 * i + 1] / (float)kH;
      o[i]     = lx * (float)kW - 0.5f;
      o[4 + i] = ly * (float)kH - 0.5f;
      o[8 + i] = e[i] / sum;
    }
  }
}

// ---------------------------------------------------------------------------
// Bilinear sampling from f16 value map; writes samp in (nq,bs,np,d) layout.
// ---------------------------------------------------------------------------
__global__ void __launch_bounds__(256) sample_kernel(
    const half_t* __restrict__ v, const float* __restrict__ locaw,
    float* __restrict__ samp) {
  const int qi = blockIdx.x;     // [0, 23040)
  const int d = threadIdx.x;     // [0, 256)
  int b = qi / (kNQ * kNP), qq = qi - b * (kNQ * kNP);
  const float* la = locaw + (long)qi * 12;
  const half_t* vb = v + (long)b * kHW * kD;
  float acc = 0.f;
#pragma unroll
  for (int pt = 0; pt < 4; ++pt) {
    float x = la[pt], y = la[4 + pt], aw = la[8 + pt];
    float x0f = floorf(x), y0f = floorf(y);
    float lx = x - x0f, ly = y - y0f;
    int x0 = (int)x0f, y0 = (int)y0f;
    float s = 0.f;
#pragma unroll
    for (int c = 0; c < 4; ++c) {
      int xi = x0 + (c & 1), yi = y0 + (c >> 1);
      float w = ((c & 1) ? lx : 1.f - lx) * ((c >> 1) ? ly : 1.f - ly);
      if (xi >= 0 && xi < kW && yi >= 0 && yi < kH)
        s += w * (float)vb[((long)yi * kW + xi) * kD + d];
    }
    acc += aw * s;
  }
  int q = qq / kNP, p = qq - q * kNP;
  samp[((long)((q * kBS + b) * kNP + p)) * kD + d] = acc;
}

// ---------------------------------------------------------------------------
// Final LayerNorm fused with output transpose:
// d_out[((b*NQ+q)*D + c)*NP + p] = LN(preln row (q*BS+b)*NP+p)
// ---------------------------------------------------------------------------
__global__ void __launch_bounds__(256) ln_out_kernel(
    const float* __restrict__ preln, const float* __restrict__ g,
    const float* __restrict__ bta, float* __restrict__ outp) {
  const int lane = threadIdx.x & 31;
  const int wave = threadIdx.x >> 5;
  long row = (long)blockIdx.x * 8 + wave;
  if (row >= (long)kTOK) return;
  int p = (int)(row % kNP);
  int t2 = (int)(row / kNP);
  int b = t2 % kBS;
  int q = t2 / kBS;
  const float* xr = preln + row * kD;
  float vals[8], s0 = 0.f, s1 = 0.f;
#pragma unroll
  for (int j = 0; j < 8; ++j) {
    float v = xr[lane * 8 + j];
    vals[j] = v; s0 += v; s1 += v * v;
  }
#pragma unroll
  for (int mk = 16; mk >= 1; mk >>= 1) {
    s0 += __shfl_xor(s0, mk, 32);
    s1 += __shfl_xor(s1, mk, 32);
  }
  float mean = s0 * (1.f / 256.f);
  float var = s1 * (1.f / 256.f) - mean * mean;
  float inv = rsqrtf(var + 1e-5f);
  float* ob = outp + ((long)(b * kNQ + q) * kD) * kNP + p;
#pragma unroll
  for (int j = 0; j < 8; ++j) {
    int c = lane * 8 + j;
    ob[(long)c * kNP] = (vals[j] - mean) * inv * g[c] + bta[c];
  }
}

// ---------------------------------------------------------------------------
// Host orchestration
// ---------------------------------------------------------------------------
extern "C" void kernel_launch(void* const* d_in, const int* in_sizes, int n_in,
                              void* d_out, int out_size, void* d_ws, size_t ws_size,
                              hipStream_t stream) {
  const float* feature   = (const float*)d_in[0];
  const float* ref_pts   = (const float*)d_in[1];
  const float* src_fl    = (const float*)d_in[2];
  const float* col_in_w  = (const float*)d_in[5];
  const float* col_in_b  = (const float*)d_in[6];
  const float* col_out_w = (const float*)d_in[7];
  const float* col_out_b = (const float*)d_in[8];
  const float* col_ln_g  = (const float*)d_in[9];
  const float* col_ln_b  = (const float*)d_in[10];
  const float* row_in_w  = (const float*)d_in[11];
  const float* row_in_b  = (const float*)d_in[12];
  const float* row_out_w = (const float*)d_in[13];
  const float* row_out_b = (const float*)d_in[14];
  const float* row_ln_g  = (const float*)d_in[15];
  const float* row_ln_b  = (const float*)d_in[16];
  const float* off_w     = (const float*)d_in[17];
  const float* off_b     = (const float*)d_in[18];
  const float* aw_w      = (const float*)d_in[19];
  const float* aw_b      = (const float*)d_in[20];
  const float* vproj_w   = (const float*)d_in[21];
  const float* vproj_b   = (const float*)d_in[22];
  const float* oproj_w   = (const float*)d_in[23];
  const float* oproj_b   = (const float*)d_in[24];
  const float* lin1_w    = (const float*)d_in[25];
  const float* lin1_b    = (const float*)d_in[26];
  const float* lin2_w    = (const float*)d_in[27];
  const float* lin2_b    = (const float*)d_in[28];
  const float* ln2_g     = (const float*)d_in[29];
  const float* ln2_b     = (const float*)d_in[30];

  // ---- workspace carve-out ----
  char* ws = (char*)d_ws;
  half_t* wf = (half_t*)ws;
  half_t* w_col_in  = wf + 0;        // 3*65536
  half_t* w_col_out = wf + 196608;   // 65536
  half_t* w_row_in  = wf + 262144;   // 3*65536
  half_t* w_row_out = wf + 458752;
  half_t* w_vproj   = wf + 524288;
  half_t* w_oproj   = wf + 589824;
  half_t* w_lin1    = wf + 655360;
  half_t* w_lin2    = wf + 720896;   // ends at 786432 halfs
  size_t off = 786432 * sizeof(half_t);
  float* tgt1  = (float*)(ws + off); off += (size_t)kTOK * kD * 4;
  float* tgt2  = (float*)(ws + off); off += (size_t)kTOK * kD * 4;
  half_t* vout = (half_t*)(ws + off); off += (size_t)kBS * kHW * kD * 2;
  float* locaw = (float*)(ws + off); off += (size_t)kTOK * 12 * 4;
  float* tgt3  = (float*)(ws + off); off += (size_t)kTOK * kD * 4;
  float* samp  = tgt1;               // reuse: tgt1 free after row attention
  half_t* hid  = (half_t*)tgt1;      // reuse: samp free after oproj
  float* preln = tgt2;               // reuse: tgt2 free after oproj residual

  // ---- convert weights to f16 ----
  const float* csrc[8] = {col_in_w, col_out_w, row_in_w, row_out_w,
                          vproj_w, oproj_w, lin1_w, lin2_w};
  half_t* cdst[8] = {w_col_in, w_col_out, w_row_in, w_row_out,
                     w_vproj, w_oproj, w_lin1, w_lin2};
  const int csz[8] = {196608, 65536, 196608, 65536, 65536, 65536, 65536, 65536};
  for (int i = 0; i < 8; ++i)
    cvt_f2h_kernel<<<(csz[i] + 255) / 256, 256, 0, stream>>>(csrc[i], cdst[i], csz[i]);

  // ---- col self-attention: seq = NP over each (nq,bs) ----
  mha_ln_kernel<kNP><<<kNQ * kBS, 256, mha_smem(kNP), stream>>>(
      feature, tgt1, w_col_in, col_in_b, w_col_out, col_out_b, col_ln_g, col_ln_b,
      (long)kD, (long)kNP * kD);
  // ---- row self-attention: seq = NQ over each (bs,np) ----
  mha_ln_kernel<kNQ><<<kBS * kNP, 256, mha_smem(kNQ), stream>>>(
      tgt1, tgt2, w_row_in, row_in_b, w_row_out, row_out_b, row_ln_g, row_ln_b,
      (long)kBS * kNP * kD, (long)kD);
  // ---- value projection (largest GEMM; f16 output halves HBM store traffic) ----
  gemm256_wmma<float, false, false, half_t><<<(kBS * kHW) / 128, 256, 0, stream>>>(
      src_fl, w_vproj, vproj_b, nullptr, vout, kBS * kHW);
  // ---- sampling locations + attention weights ----
  offaw_kernel<<<kTOK / 8, 256, 0, stream>>>(tgt2, ref_pts, off_w, off_b,
                                             aw_w, aw_b, locaw);
  // ---- bilinear gather ----
  sample_kernel<<<kTOK, 256, 0, stream>>>(vout, locaw, samp);
  // ---- output projection + residual ----
  gemm256_wmma<float, false, true, float><<<kTOK / 128, 256, 0, stream>>>(
      samp, w_oproj, oproj_b, tgt2, tgt3, kTOK);
  // ---- FFN ----
  gemm256_wmma<float, true, false, half_t><<<kTOK / 128, 256, 0, stream>>>(
      tgt3, w_lin1, lin1_b, nullptr, hid, kTOK);
  gemm256_wmma<half_t, false, true, float><<<kTOK / 128, 256, 0, stream>>>(
      hid, w_lin2, lin2_b, tgt3, preln, kTOK);
  // ---- final LN + transpose into d_out ----
  ln_out_kernel<<<kTOK / 8, 256, 0, stream>>>(preln, ln2_g, ln2_b, (float*)d_out);

  (void)in_sizes; (void)n_in; (void)out_size; (void)ws_size;
}
